// RNNScratch_38362647888375
// MI455X (gfx1250) — compile-verified, matrix operand order checked by reading, not run on hardware
//
#include <hip/hip_runtime.h>
#include <math.h>

// Problem constants (match reference: T, B, I, H)
#define RT 512
#define RB 64
#define RI 1024
#define RH 1024

typedef __attribute__((ext_vector_type(16))) __bf16 v16bf;
typedef __attribute__((ext_vector_type(8)))  __bf16 v8bf;
typedef __attribute__((ext_vector_type(8)))  float  v8f;

// ---------------------------------------------------------------------------
// WMMA fragment loaders (CDNA5 wave32, V_WMMA_F32_16X16X32_BF16 layouts)
//
// A-fragment (16x32 bf16, ISA 7.12.2): lane l -> row M = l%16; hi = l/16.
//   frag elements e=0..7  hold K = kBase + 8*hi + e
//   frag elements e=8..15 hold K = kBase + 16 + 8*hi + (e-8)
// B-fragment: same pattern with N = l%16, sourced from W^T stored [N][K]
// row-major, so every lane reads two contiguous 16-byte chunks.
// ---------------------------------------------------------------------------

__device__ __forceinline__ v16bf load_frag_bf16(const __bf16* __restrict__ base,
                                                int ldk, int lane) {
  const int r  = lane & 15;
  const int hi = lane >> 4;
  const __bf16* p = base + (size_t)r * ldk;
  v8bf lo = *(const v8bf*)(p + 8 * hi);
  v8bf hh = *(const v8bf*)(p + 16 + 8 * hi);
  v16bf f;
#pragma unroll
  for (int i = 0; i < 8; ++i) { f[i] = lo[i]; f[8 + i] = hh[i]; }
  return f;
}

__device__ __forceinline__ v16bf load_frag_f32_cvt(const float* __restrict__ base,
                                                   int ldk, int lane) {
  const int r  = lane & 15;
  const int hi = lane >> 4;
  const float* p = base + (size_t)r * ldk;
  v8f lo = *(const v8f*)(p + 8 * hi);
  v8f hh = *(const v8f*)(p + 16 + 8 * hi);
  v16bf f;
#pragma unroll
  for (int i = 0; i < 8; ++i) { f[i] = (__bf16)lo[i]; f[8 + i] = (__bf16)hh[i]; }
  return f;
}

// ---------------------------------------------------------------------------
// Phase 0: convert W_xh [I][H] and W_hh [H][H] (fp32, row-major, K-major) to
// bf16 transposed [N][K] layouts in workspace; zero bf16 state ping-pong bufs.
// ---------------------------------------------------------------------------
__global__ void rnn_prep_kernel(const float* __restrict__ Wxh,
                                const float* __restrict__ Whh,
                                __bf16* __restrict__ WxhT,
                                __bf16* __restrict__ WhhT,
                                __bf16* __restrict__ st0,
                                __bf16* __restrict__ st1) {
  size_t idx = (size_t)blockIdx.x * blockDim.x + threadIdx.x;  // over I*H = 1M
  if (idx < (size_t)RI * RH) {
    int k = (int)(idx / RH);   // source row   (K dimension)
    int n = (int)(idx % RH);   // source col   (N dimension) -> coalesced reads
    WxhT[(size_t)n * RI + k] = (__bf16)Wxh[idx];
    WhhT[(size_t)n * RH + k] = (__bf16)Whh[idx];
  }
  if (idx < (size_t)RB * RH) {
    st0[idx] = (__bf16)0.0f;
    st1[idx] = (__bf16)0.0f;
  }
}

// ---------------------------------------------------------------------------
// Phase 1: Z = X @ W_xh + b  ->  written straight into d_out[t*B*H ...].
// X is [T*B][I] fp32 row-major (T,B,I contiguous). One wave computes a
// 16x64 output strip (A fragment reused across 4 WMMAs per K-step).
// waves = (T*B/16) * (H/64) = 2048 * 16 = 32768.
// ---------------------------------------------------------------------------
__global__ __launch_bounds__(256) void rnn_xgemm_kernel(
    const float* __restrict__ X, const __bf16* __restrict__ WxhT,
    const float* __restrict__ bias, float* __restrict__ Z) {
  const int wave = (int)((blockIdx.x * blockDim.x + threadIdx.x) >> 5);
  const int lane = (int)(threadIdx.x & 31);
  const int mTile  = wave >> 4;        // 0..2047
  const int nStrip = wave & 15;        // 0..15
  const int m0 = mTile * 16;
  const int n0 = nStrip * 64;
  const int r  = lane & 15;
  const int hi = lane >> 4;

  v8f c0 = {}, c1 = {}, c2 = {}, c3 = {};
  for (int k = 0; k < RI; k += 32) {
    v16bf a = load_frag_f32_cvt(X + (size_t)m0 * RI + k, RI, lane);
    v16bf b0 = load_frag_bf16(WxhT + (size_t)(n0 +  0) * RI + k, RI, lane);
    v16bf b1 = load_frag_bf16(WxhT + (size_t)(n0 + 16) * RI + k, RI, lane);
    v16bf b2 = load_frag_bf16(WxhT + (size_t)(n0 + 32) * RI + k, RI, lane);
    v16bf b3 = load_frag_bf16(WxhT + (size_t)(n0 + 48) * RI + k, RI, lane);
    c0 = __builtin_amdgcn_wmma_f32_16x16x32_bf16(false, a, false, b0, (short)0, c0, false, false);
    c1 = __builtin_amdgcn_wmma_f32_16x16x32_bf16(false, a, false, b1, (short)0, c1, false, false);
    c2 = __builtin_amdgcn_wmma_f32_16x16x32_bf16(false, a, false, b2, (short)0, c2, false, false);
    c3 = __builtin_amdgcn_wmma_f32_16x16x32_bf16(false, a, false, b3, (short)0, c3, false, false);
  }

  // D-fragment element v lives at (row m0 + v + 8*hi, col n0 + 16*j + r)
#pragma unroll
  for (int j = 0; j < 4; ++j) {
    const v8f c = (j == 0) ? c0 : (j == 1) ? c1 : (j == 2) ? c2 : c3;
    const int n = n0 + 16 * j + r;
    const float bv = bias[n];
#pragma unroll
    for (int v = 0; v < 8; ++v) {
      Z[(size_t)(m0 + v + 8 * hi) * RH + n] = c[v] + bv;
    }
  }
}

// ---------------------------------------------------------------------------
// Phase 2 (one launch per timestep): h_t = tanh(Z_t + h_{t-1} @ W_hh).
// Z tile is read from d_out as the fp32 accumulator, 32 chained bf16 WMMAs
// add the recurrent term, tanh is applied, fp32 result overwrites d_out and
// a bf16 copy goes to the next-state ping-pong buffer.
// 256 tiles (4 M-tiles x 64 N-tiles), one wave each: 32 blocks x 256 thr.
// ---------------------------------------------------------------------------
__global__ __launch_bounds__(256) void rnn_step_kernel(
    const __bf16* __restrict__ WhhT, const __bf16* __restrict__ state_in,
    __bf16* __restrict__ state_out, float* __restrict__ z_inout) {
  const int wave = (int)((blockIdx.x * blockDim.x + threadIdx.x) >> 5);
  const int lane = (int)(threadIdx.x & 31);
  const int mTile = wave >> 6;   // 0..3
  const int nTile = wave & 63;   // 0..63
  const int m0 = mTile * 16;
  const int n0 = nTile * 16;
  const int r  = lane & 15;
  const int hi = lane >> 4;
  const int n  = n0 + r;

  // Seed accumulator with pre-activation tile from d_out.
  v8f c;
#pragma unroll
  for (int v = 0; v < 8; ++v) {
    c[v] = z_inout[(size_t)(m0 + v + 8 * hi) * RH + n];
  }

  for (int k = 0; k < RH; k += 32) {
    v16bf a = load_frag_bf16(state_in + (size_t)m0 * RH + k, RH, lane);
    v16bf b = load_frag_bf16(WhhT + (size_t)n0 * RH + k, RH, lane);
    c = __builtin_amdgcn_wmma_f32_16x16x32_bf16(false, a, false, b, (short)0, c, false, false);
  }

#pragma unroll
  for (int v = 0; v < 8; ++v) {
    const float h = tanhf(c[v]);
    const size_t off = (size_t)(m0 + v + 8 * hi) * RH + n;
    z_inout[off]   = h;           // fp32 output (d_out, in place)
    state_out[off] = (__bf16)h;   // bf16 carry for next step
  }
}

// ---------------------------------------------------------------------------
// Final: copy outputs[T-1] into the final-state slot of d_out.
// ---------------------------------------------------------------------------
__global__ void rnn_final_copy_kernel(const float* __restrict__ src,
                                      float* __restrict__ dst, int n) {
  int i = (int)(blockIdx.x * blockDim.x + threadIdx.x);
  if (i < n) dst[i] = src[i];
}

extern "C" void kernel_launch(void* const* d_in, const int* in_sizes, int n_in,
                              void* d_out, int out_size, void* d_ws, size_t ws_size,
                              hipStream_t stream) {
  (void)in_sizes; (void)n_in; (void)out_size; (void)ws_size;

  const float* X   = (const float*)d_in[0];  // [T,B,I]
  const float* Wxh = (const float*)d_in[1];  // [I,H]
  const float* Whh = (const float*)d_in[2];  // [H,H]
  const float* bh  = (const float*)d_in[3];  // [H]
  float* out = (float*)d_out;                // [T*B*H] outputs ++ [B*H] final

  // Workspace layout (bf16): WxhT [H][I], WhhT [H][H], state ping-pong x2.
  __bf16* WxhT = (__bf16*)d_ws;
  __bf16* WhhT = WxhT + (size_t)RI * RH;
  __bf16* st0  = WhhT + (size_t)RH * RH;
  __bf16* st1  = st0  + (size_t)RB * RH;

  // Phase 0: weight convert/transpose + zero initial state. 1M threads.
  rnn_prep_kernel<<<(RI * RH) / 256, 256, 0, stream>>>(Wxh, Whh, WxhT, WhhT, st0, st1);

  // Phase 1: time-parallel X @ W_xh + b  -> d_out (pre-activations).
  // 32768 waves = 4096 blocks x 256 threads.
  rnn_xgemm_kernel<<<4096, 256, 0, stream>>>(X, WxhT, bh, out);

  // Phase 2: 512 sequential recurrent steps (stream order is the sync).
  for (int t = 0; t < RT; ++t) {
    const __bf16* sin  = (t & 1) ? st1 : st0;
    __bf16*       sout = (t & 1) ? st0 : st1;
    rnn_step_kernel<<<32, 256, 0, stream>>>(WhhT, sin, sout,
                                            out + (size_t)t * RB * RH);
  }

  // Final state = outputs[T-1].
  rnn_final_copy_kernel<<<(RB * RH + 255) / 256, 256, 0, stream>>>(
      out + (size_t)(RT - 1) * RB * RH, out + (size_t)RT * RB * RH, RB * RH);
}
